// GAL_84945863180509
// MI455X (gfx1250) — compile-verified, matrix operand order checked by reading, not run on hardware
//
#include <hip/hip_runtime.h>
#include <hip/hip_bf16.h>

// ---------------------------------------------------------------------------
// MI455X (gfx1250) implementation.
// Matrix ops use V_WMMA_F32_16X16X32_F16 (f16 A/B, f32 accumulate) — the
// precision point CDNA5 rewards (8x K-depth vs the f32 WMMA) while BN keeps
// activations ~unit scale so f16 inputs are numerically fine. BN statistics
// are accumulated in f32 from the f32 WMMA accumulators; the global std uses
// f64 accumulation. Intermediate activations live in HBM as f16 to halve
// traffic; the pipeline is memory-bound (~2 GB @ 23.3 TB/s).
// ---------------------------------------------------------------------------

typedef __attribute__((ext_vector_type(16))) _Float16 v16h;
typedef __attribute__((ext_vector_type(8)))  float    v8f;

#define BN_EPS 1e-5f

// Build a 16-bit WMMA fragment from LDS per ISA 7.12.2:
//   lanes 0-15 : K-chunks {k0..k0+7, k0+16..k0+23}
//   lanes 16-31: K-chunks {k0+8..k0+15, k0+24..k0+31}
// p points at (row/col base) + k0 + (lane<16 ? 0 : 8).
static __device__ __forceinline__ v16h load_frag(const _Float16* p) {
  v16h r;
  *(uint4*)&r         = *(const uint4*)p;        // 8 halfs, 16B
  *(((uint4*)&r) + 1) = *(const uint4*)(p + 16); // next K-chunk, 16B
  return r;
}

// ---------------- global std of (x_j - x_i) over all E*67 elements ---------
__global__ __launch_bounds__(256) void k_dstats(
    const float* __restrict__ x, const float* __restrict__ pos,
    const long long* __restrict__ ei, long long E_, double* __restrict__ acc)
{
  double s = 0.0, ss = 0.0;
  for (long long e = (long long)blockIdx.x * 256 + threadIdx.x; e < E_;
       e += (long long)gridDim.x * 256) {
    long long i = ei[e], j = ei[E_ + e];
    const float* xi = x + i * 64;  const float* xj = x + j * 64;
    for (int c = 0; c < 64; ++c) { double d = (double)(xj[c] - xi[c]); s += d; ss += d * d; }
    const float* pi = pos + i * 3; const float* pj = pos + j * 3;
    for (int c = 0; c < 3;  ++c) { double d = (double)(pj[c] - pi[c]); s += d; ss += d * d; }
  }
  __shared__ double rs[256], rss[256];
  int t = threadIdx.x;
  rs[t] = s; rss[t] = ss; __syncthreads();
  for (int o = 128; o > 0; o >>= 1) {
    if (t < o) { rs[t] += rs[t + o]; rss[t] += rss[t + o]; }
    __syncthreads();
  }
  if (t == 0) { atomicAdd(&acc[0], rs[0]); atomicAdd(&acc[1], rss[0]); }
}

__global__ void k_dfin(const double* __restrict__ acc, float* __restrict__ istd, double M)
{
  double mean = acc[0] / M;
  double var  = (acc[1] - acc[0] * mean) / (M - 1.0);   // unbiased (torch.std)
  if (var < 0.0) var = 0.0;
  istd[0] = (float)(1.0 / (sqrt(var) + (double)BN_EPS));
}

// -------- shared epilogue: bias, BN-stat accumulation, coalesced store -----
static __device__ __forceinline__ void gemm_epilogue(
    v8f* acc, _Float16* Aw /* 16x64 LDS scratch */, float* sStat,
    const float* sBias, _Float16* __restrict__ out,
    long long e0, long long E_, int lane)
{
  const int rA = lane & 15;
  const int mbase = (lane < 16) ? 0 : 8;     // C/D layout: VGPR r -> M=r / M=8+r
  for (int nt = 0; nt < 4; ++nt) {
    int n = nt * 16 + rA;                    // lane's output channel in this tile
    float bn = sBias[n];
    float s = 0.f, ss = 0.f;
    for (int i = 0; i < 8; ++i) {
      float v = acc[nt][i] + bn;
      if (e0 + mbase + i < E_) { s += v; ss += v * v; }
      Aw[(mbase + i) * 64 + n] = (_Float16)v;
    }
    atomicAdd(&sStat[n],      s);
    atomicAdd(&sStat[64 + n], ss);
  }
  // transpose through LDS -> coalesced b128 global stores
  int r = lane >> 1, h = lane & 1;
  long long e = e0 + r;
  if (e < E_) {
    const uint4* s4 = (const uint4*)&Aw[r * 64 + h * 32];
    uint4*       d4 = (uint4*)(out + e * 64 + h * 32);
    for (int q = 0; q < 4; ++q) d4[q] = s4[q];
  }
}

// ---------------- layer 1: gather + affine-normalize + GEMM (K=134->160) ---
__global__ __launch_bounds__(128) void k_gemm_edge(
    const float* __restrict__ x, const float* __restrict__ pos,
    const long long* __restrict__ ei,
    const float* __restrict__ aw, const float* __restrict__ ab,
    const float* __restrict__ W /*[134][64]*/, const float* __restrict__ bias,
    const float* __restrict__ istd_p,
    _Float16* __restrict__ out, float* __restrict__ stats, long long E_)
{
  __shared__ _Float16 Bt[64 * 160];       // weights transposed [n][k], zero-padded K
  __shared__ _Float16 At[4][16 * 160];    // per-wave activation tile [m][k]
  __shared__ float    sBias[64];
  __shared__ float    sStat[128];
  const int tid = threadIdx.x, wave = tid >> 5, lane = tid & 31;

  for (int i = tid; i < 64 * 160; i += 128) {
    int n = i / 160, k = i % 160;
    Bt[i] = (k < 134) ? (_Float16)W[k * 64 + n] : (_Float16)0.f;
  }
  if (tid < 64)  sBias[tid] = bias[tid];
  if (tid < 128) sStat[tid] = 0.f;

  const float istd = istd_p[0];
  long long e0 = (long long)blockIdx.x * 64 + (long long)wave * 16;

  for (int r = 0; r < 16; ++r) {
    long long e = e0 + r; if (e >= E_) e = E_ - 1;
    long long i = ei[e], j = ei[E_ + e];
    for (int c = lane; c < 160; c += 32) {
      float v;
      if (c < 67) {
        v = (c < 64) ? x[i * 64 + c] : pos[i * 3 + (c - 64)];
      } else if (c < 134) {
        int cc = c - 67;
        float fi = (cc < 64) ? x[i * 64 + cc] : pos[i * 3 + (cc - 64)];
        float fj = (cc < 64) ? x[j * 64 + cc] : pos[j * 3 + (cc - 64)];
        v = aw[cc] * ((fj - fi) * istd) + ab[cc];
      } else v = 0.f;                       // K padding
      At[wave][r * 160 + c] = (_Float16)v;
    }
  }
  __syncthreads();

  const int hsel = (lane < 16) ? 0 : 8;
  const int rA = lane & 15;
  v8f acc[4];
  for (int i = 0; i < 4; ++i) for (int q = 0; q < 8; ++q) acc[i][q] = 0.f;

  for (int ks = 0; ks < 5; ++ks) {
    const int kb = ks * 32;
    v16h a = load_frag(&At[wave][rA * 160 + kb + hsel]);
    for (int nt = 0; nt < 4; ++nt) {
      v16h b = load_frag(&Bt[(nt * 16 + rA) * 160 + kb + hsel]);
      acc[nt] = __builtin_amdgcn_wmma_f32_16x16x32_f16(
          false, a, false, b, (short)0, acc[nt], false, false);
    }
  }
  gemm_epilogue(acc, &At[wave][0], sStat, sBias, out, e0, E_, lane);
  __syncthreads();
  if (tid < 128) atomicAdd(&stats[tid], sStat[tid]);
}

// ---------------- 64x64 GEMM, optional fused BN+ReLU on input --------------
__global__ __launch_bounds__(128) void k_gemm64(
    const _Float16* __restrict__ in, const float* __restrict__ sb /* [scale64,bias64] or null */,
    const float* __restrict__ W /*[64][64]*/, const float* __restrict__ bias,
    _Float16* __restrict__ out, float* __restrict__ stats, long long E_)
{
  __shared__ _Float16 Bt[64 * 64];        // [n][k]
  __shared__ _Float16 At[4][16 * 64];     // per-wave [m][k]
  __shared__ float    sBias[64];
  __shared__ float    sStat[128];
  const int tid = threadIdx.x, wave = tid >> 5, lane = tid & 31;

  for (int i = tid; i < 64 * 64; i += 128) {
    int k = i >> 6, n = i & 63;
    Bt[n * 64 + k] = (_Float16)W[i];
  }
  if (tid < 64)  sBias[tid] = bias[tid];
  if (tid < 128) sStat[tid] = 0.f;

  long long e0 = (long long)blockIdx.x * 64 + (long long)wave * 16;

  // stage A tile (16 rows x 64 cols), fusing BN+ReLU if requested
  {
    int r = lane >> 1, h = lane & 1;
    long long e = e0 + r; if (e >= E_) e = E_ - 1;
    const uint4* src = (const uint4*)(in + e * 64 + (long long)h * 32);
    union { uint4 u; _Float16 hh[8]; } b;
    for (int q = 0; q < 4; ++q) {
      b.u = src[q];
      if (sb) {
        for (int t2 = 0; t2 < 8; ++t2) {
          int ch = h * 32 + q * 8 + t2;
          float v = (float)b.hh[t2];
          b.hh[t2] = (_Float16)fmaxf(sb[ch] * v + sb[64 + ch], 0.f);
        }
      }
      *(uint4*)&At[wave][r * 64 + h * 32 + q * 8] = b.u;
    }
  }
  __syncthreads();

  const int hsel = (lane < 16) ? 0 : 8;
  const int rA = lane & 15;
  v8f acc[4];
  for (int i = 0; i < 4; ++i) for (int q = 0; q < 8; ++q) acc[i][q] = 0.f;

  for (int ks = 0; ks < 2; ++ks) {
    const int kb = ks * 32;
    v16h a = load_frag(&At[wave][rA * 64 + kb + hsel]);
    for (int nt = 0; nt < 4; ++nt) {
      v16h b = load_frag(&Bt[(nt * 16 + rA) * 64 + kb + hsel]);
      acc[nt] = __builtin_amdgcn_wmma_f32_16x16x32_f16(
          false, a, false, b, (short)0, acc[nt], false, false);
    }
  }
  gemm_epilogue(acc, &At[wave][0], sStat, sBias, out, e0, E_, lane);
  __syncthreads();
  if (tid < 128) atomicAdd(&stats[tid], sStat[tid]);
}

// ---------------- BN stat finalize: fold into per-channel scale/bias -------
__global__ void k_bnfin(const float* __restrict__ stats, const float* __restrict__ g,
                        const float* __restrict__ be, float* __restrict__ sb, float invM)
{
  int c = threadIdx.x;
  if (c < 64) {
    float mean = stats[c] * invM;
    float var  = stats[64 + c] * invM - mean * mean;   // biased (BN training)
    float s    = g[c] * rsqrtf(var + BN_EPS);
    sb[c]      = s;
    sb[64 + c] = be[c] - mean * s;
  }
}

// ---------------- h = relu(s*y + b) ----------------------------------------
__global__ __launch_bounds__(256) void k_apply(
    const _Float16* __restrict__ y, const float* __restrict__ sb,
    _Float16* __restrict__ h, long long total8)
{
  long long t = (long long)blockIdx.x * 256 + threadIdx.x;
  if (t >= total8) return;
  long long base = t * 8;
  union { uint4 u; _Float16 hh[8]; } b;
  b.u = *(const uint4*)(y + base);
  int c0 = (int)(base & 63);
  for (int i = 0; i < 8; ++i) {
    int ch = c0 + i;
    float v = (float)b.hh[i];
    b.hh[i] = (_Float16)fmaxf(sb[ch] * v + sb[64 + ch], 0.f);
  }
  *(uint4*)(h + base) = b.u;
}

// ---------------- h = relu(s*t + b + h) (residual) -------------------------
__global__ __launch_bounds__(256) void k_resid(
    const _Float16* __restrict__ tbuf, const float* __restrict__ sb,
    _Float16* __restrict__ h, long long total8)
{
  long long t = (long long)blockIdx.x * 256 + threadIdx.x;
  if (t >= total8) return;
  long long base = t * 8;
  union { uint4 u; _Float16 hh[8]; } bt, bh;
  bt.u = *(const uint4*)(tbuf + base);
  bh.u = *(const uint4*)(h + base);
  int c0 = (int)(base & 63);
  for (int i = 0; i < 8; ++i) {
    int ch = c0 + i;
    float v = sb[ch] * (float)bt.hh[i] + sb[64 + ch] + (float)bh.hh[i];
    bh.hh[i] = (_Float16)fmaxf(v, 0.f);
  }
  *(uint4*)(h + base) = bh.u;
}

// ---------------- segment max/sum/count via atomics ------------------------
// Post-ReLU values are >= 0, so uint-bit atomicMax is order-correct and a
// zero-initialized max buffer exactly reproduces the empty-segment -> 0 rule.
__global__ __launch_bounds__(256) void k_scatter(
    const _Float16* __restrict__ hbuf, const long long* __restrict__ ei,
    unsigned* __restrict__ mx, float* __restrict__ sm, float* __restrict__ cnt,
    long long E_)
{
  long long t = (long long)blockIdx.x * 256 + threadIdx.x;
  long long e = t >> 3; int q = (int)(t & 7);
  if (e >= E_) return;
  long long n = ei[e];                 // idx_i
  union { uint4 u; _Float16 hh[8]; } b;
  b.u = *(const uint4*)(hbuf + e * 64 + (long long)q * 8);
  for (int i = 0; i < 8; ++i) {
    float v = (float)b.hh[i];
    long long o = n * 64 + q * 8 + i;
    atomicMax(&mx[o], __float_as_uint(v));
    atomicAdd(&sm[o], v);
  }
  if (q == 0) atomicAdd(&cnt[n], 1.f);
}

__global__ __launch_bounds__(256) void k_final(
    const unsigned* __restrict__ mx, const float* __restrict__ sm,
    const float* __restrict__ cnt, float* __restrict__ out, long long total)
{
  long long t = (long long)blockIdx.x * 256 + threadIdx.x;
  if (t >= total) return;
  long long n = t >> 6;
  float c = cnt[n]; if (c < 1.f) c = 1.f;
  out[t] = __uint_as_float(mx[t]) + sm[t] / c;
}

// ---------------------------------------------------------------------------
extern "C" void kernel_launch(void* const* d_in, const int* in_sizes, int n_in,
                              void* d_out, int out_size, void* d_ws, size_t ws_size,
                              hipStream_t stream)
{
  (void)n_in; (void)out_size; (void)ws_size;
  const float*     pos  = (const float*)d_in[0];
  const float*     x    = (const float*)d_in[1];
  const long long* ei   = (const long long*)d_in[2];   // jnp.int64 per reference
  const float*     aw   = (const float*)d_in[3];
  const float*     ab   = (const float*)d_in[4];
  const float*     linW = (const float*)d_in[5];
  const float*     linb = (const float*)d_in[6];
  const float*     ling = (const float*)d_in[7];
  const float*     linbe= (const float*)d_in[8];
  const float*     rW1  = (const float*)d_in[9];
  const float*     rb1  = (const float*)d_in[10];
  const float*     rg1  = (const float*)d_in[11];
  const float*     rbe1 = (const float*)d_in[12];
  const float*     rW2  = (const float*)d_in[13];
  const float*     rb2  = (const float*)d_in[14];
  const float*     rg2  = (const float*)d_in[15];
  const float*     rbe2 = (const float*)d_in[16];

  const long long Nn = (long long)in_sizes[0] / 3;
  const long long E_ = (long long)in_sizes[2] / 2;

  // workspace layout (zero-init region first, then f16 activation buffers)
  char* ws = (char*)d_ws;
  size_t off = 0;
  auto alloc = [&](size_t bytes) {
    size_t o = off; off = (off + bytes + 255) & ~(size_t)255; return o;
  };
  size_t o_mx    = alloc((size_t)Nn * 64 * 4);
  size_t o_sm    = alloc((size_t)Nn * 64 * 4);
  size_t o_cnt   = alloc((size_t)Nn * 4);
  size_t o_dacc  = alloc(2 * sizeof(double));
  size_t o_istd  = alloc(16);
  size_t o_stats = alloc(5 * 128 * 4);
  size_t zero_end = off;
  size_t o_sb    = alloc(5 * 128 * 4);
  size_t o_bufA  = alloc((size_t)E_ * 64 * 2);
  size_t o_bufB  = alloc((size_t)E_ * 64 * 2);

  unsigned*  mx    = (unsigned*)(ws + o_mx);
  float*     sm    = (float*)(ws + o_sm);
  float*     cnt   = (float*)(ws + o_cnt);
  double*    dacc  = (double*)(ws + o_dacc);
  float*     istd  = (float*)(ws + o_istd);
  float*     stats = (float*)(ws + o_stats);
  float*     sb    = (float*)(ws + o_sb);
  _Float16*  bufA  = (_Float16*)(ws + o_bufA);
  _Float16*  bufB  = (_Float16*)(ws + o_bufB);

  hipMemsetAsync(d_ws, 0, zero_end, stream);

  // global std of feature diffs
  k_dstats<<<1024, 256, 0, stream>>>(x, pos, ei, E_, dacc);
  k_dfin<<<1, 1, 0, stream>>>(dacc, istd, (double)E_ * 67.0);

  const int gb = (int)((E_ + 63) / 64);          // 64 edges per block (4 waves)
  const float invM = 1.f / (float)E_;
  const long long tot8 = E_ * 64 / 8;
  const int ga = (int)((tot8 + 255) / 256);

  // shared_lin: Linear(134->64) + BN + ReLU
  k_gemm_edge<<<gb, 128, 0, stream>>>(x, pos, ei, aw, ab, linW, linb, istd,
                                      bufA, stats + 0 * 128, E_);
  k_bnfin<<<1, 64, 0, stream>>>(stats + 0 * 128, ling, linbe, sb + 0 * 128, invM);
  k_apply<<<ga, 256, 0, stream>>>(bufA, sb + 0 * 128, bufB, tot8);  // bufB = h

  // ResMLP blocks
  for (int k = 0; k < 2; ++k) {
    float* st1 = stats + (1 + 2 * k) * 128;
    float* st2 = stats + (2 + 2 * k) * 128;
    float* sb1 = sb    + (1 + 2 * k) * 128;
    float* sb2 = sb    + (2 + 2 * k) * 128;
    k_gemm64<<<gb, 128, 0, stream>>>(bufB, nullptr, rW1 + k * 4096, rb1 + k * 64,
                                     bufA, st1, E_);                 // u = h@W1+b1
    k_bnfin<<<1, 64, 0, stream>>>(st1, rg1 + k * 64, rbe1 + k * 64, sb1, invM);
    k_gemm64<<<gb, 128, 0, stream>>>(bufA, sb1, rW2 + k * 4096, rb2 + k * 64,
                                     bufA, st2, E_);                 // t = relu(bn(u))@W2+b2 (in-place, tile-local)
    k_bnfin<<<1, 64, 0, stream>>>(st2, rg2 + k * 64, rbe2 + k * 64, sb2, invM);
    k_resid<<<ga, 256, 0, stream>>>(bufA, sb2, bufB, tot8);          // h = relu(bn(t)+h)
  }

  // dual scatter reduce + combine
  const long long st_tot = E_ * 8;
  k_scatter<<<(int)((st_tot + 255) / 256), 256, 0, stream>>>(bufB, ei, mx, sm, cnt, E_);
  const long long ft = Nn * 64;
  k_final<<<(int)((ft + 255) / 256), 256, 0, stream>>>(mx, sm, cnt, (float*)d_out, ft);
}
